// scMSAC_34222299415043
// MI455X (gfx1250) — compile-verified, matrix operand order checked by reading, not run on hardware
//
#include <hip/hip_runtime.h>

#define DEV __device__ __forceinline__

typedef __attribute__((ext_vector_type(16))) __bf16        v16bf;
typedef __attribute__((ext_vector_type(8)))  float         v8f;
typedef __attribute__((ext_vector_type(8)))  unsigned int  v8u;
typedef __attribute__((ext_vector_type(4)))  unsigned int  v4u;
typedef __attribute__((ext_vector_type(8)))  int           v8i;
typedef __attribute__((ext_vector_type(4)))  int           v4i;

static constexpr int   NN     = 16000;
static constexpr float SIGMA1 = 2.5f;
static constexpr float THO_C  = 1.0f;
static constexpr float BN_EPS = 1e-5f;
static constexpr float LN_EPS = 1e-6f;

// ---------------- device helpers ----------------

DEV unsigned short f2bf(float f) {
    unsigned int u = __float_as_uint(f);
    u += 0x7FFFu + ((u >> 16) & 1u);          // round-to-nearest-even
    return (unsigned short)(u >> 16);
}
DEV int   fOrd(float f)  { int i = __float_as_int(f); return i >= 0 ? i : (i ^ 0x7FFFFFFF); }
DEV float fUnord(int i)  { return __int_as_float(i >= 0 ? i : (i ^ 0x7FFFFFFF)); }
DEV float wave_sum32(float v) {
    for (int o = 16; o; o >>= 1) v += __shfl_xor(v, o, 32);
    return v;
}
DEV v8f wmma_bf16(v16bf a, v16bf b, v8f c) {
    return __builtin_amdgcn_wmma_f32_16x16x32_bf16(false, a, false, b, (short)0, c,
                                                   false, false);
}

// Tensor Data Mover: stage a 32-row x KC-col bf16 tile (row stride Kpad elems)
// from global BT into LDS at byte offset lds_byte. D# built per CDNA5 ISA 8.3/8.4.
DEV void tdm_load_b(const unsigned short* gptr, unsigned lds_byte, int KC, int Kpad) {
    unsigned long long ga = (unsigned long long)gptr;
    v4u g0;
    g0.x = 1u;                                          // count=1, user descriptor
    g0.y = lds_byte;                                    // lds_addr
    g0.z = (unsigned)(ga & 0xFFFFFFFFu);                // global_addr[31:0]
    g0.w = (unsigned)((ga >> 32) & 0x01FFFFFFu) | (2u << 30); // addr[56:32] | type=2
    v8i g1;
    g1[0] = (int)(1u << 16);                            // data_size=2B, no flags
    g1[1] = (int)(((unsigned)KC & 0xFFFFu) << 16);      // abar=0 | tensor_dim0.lo16
    g1[2] = (int)((((unsigned)KC >> 16) & 0xFFFFu) | (32u << 16)); // dim0.hi | dim1.lo=32
    g1[3] = (int)(((unsigned)KC & 0xFFFFu) << 16);      // dim1.hi=0 | tile_dim0=KC
    g1[4] = 32;                                         // tile_dim1=32, tile_dim2=0
    g1[5] = Kpad;                                       // tensor_dim0_stride.lo32
    g1[6] = 0;                                          // stride0.hi16 | stride1.lo16
    g1[7] = 0;                                          // stride1.hi32
    v4i z4 = {0, 0, 0, 0};
#if __clang_major__ >= 23
    v8i z8 = {0, 0, 0, 0, 0, 0, 0, 0};
    __builtin_amdgcn_tensor_load_to_lds(g0, g1, z4, z4, z8, 0);
#else
    __builtin_amdgcn_tensor_load_to_lds(g0, g1, z4, z4, 0);
#endif
}

// ---------------- utility kernels ----------------

__global__ void k_fill_f32(float* __restrict__ p, float v, long n) {
    long i = (long)blockIdx.x * blockDim.x + threadIdx.x;
    if (i < n) p[i] = v;
}
__global__ void k_fill_i32(int* __restrict__ p, int v, long n) {
    long i = (long)blockIdx.x * blockDim.x + threadIdx.x;
    if (i < n) p[i] = v;
}

// fp32 [N,K] (+ optional noise*sigma) -> bf16 [N,Kpad] zero-padded
__global__ void k_cvt_x(const float* __restrict__ x, const float* __restrict__ noise,
                        float sigma, unsigned short* __restrict__ out,
                        int N, int K, int Kpad) {
    long i = (long)blockIdx.x * blockDim.x + threadIdx.x;
    long total = (long)N * Kpad;
    if (i >= total) return;
    int n = (int)(i / Kpad), k = (int)(i % Kpad);
    float v = 0.f;
    if (k < K) {
        v = x[(size_t)n * K + k];
        if (noise) v += noise[(size_t)n * K + k] * sigma;
    }
    out[i] = f2bf(v);
}

// fp32 weight [K,Nout] -> bf16 transposed [NpadRows,Kpad], zero-padded both dims
__global__ void k_cvt_wT(const float* __restrict__ w, unsigned short* __restrict__ out,
                         int K, int Nout, int Kpad, int NpadRows) {
    long i = (long)blockIdx.x * blockDim.x + threadIdx.x;
    long total = (long)NpadRows * Kpad;
    if (i >= total) return;
    int n = (int)(i / Kpad), k = (int)(i % Kpad);
    float v = (k < K && n < Nout) ? w[(size_t)k * Nout + n] : 0.f;
    out[i] = f2bf(v);
}

// ---------------- WMMA bf16 GEMM with TDM-staged B ----------------
// C[M,Nout] = A[M,Kpad] * BT[Npad32,Kpad]^T.
// Block: 4 waves, 128 M-rows x 32 N-cols. Each wave computes a 32x32 tile
// (2x2 x 16x16, 4 WMMAs/K-step). B slab (32 rows x KC) is staged into LDS by
// the Tensor Data Mover, double-buffered across K-chunks, synced with
// s_wait_tensorcnt + workgroup barrier. A streams from global (b128 + prefetch).
// Requires: M % 128 == 0, Kpad == KC * nChunk, KC <= 256, Nout % 16 == 0.
// act: 0=none 1=ELU 2=clip(exp) 3=clip(softplus) 4=sigmoid; bn before act.
__global__ void __launch_bounds__(128)
k_gemm(const unsigned short* __restrict__ A,
       const unsigned short* __restrict__ BT,
       float* __restrict__ C,
       int Nout, int Kpad, int KC, int nChunk,
       const float* __restrict__ bias,
       const float* __restrict__ bnG, const float* __restrict__ bnB,
       const float* __restrict__ bnM, const float* __restrict__ bnV,
       int act) {
    __shared__ __align__(64) unsigned short sB[2][32 * 256];

    const int lane = threadIdx.x & 31;
    const int wave = threadIdx.x >> 5;
    const int ng   = blockIdx.x;                         // 32-col group
    const int mrow = (blockIdx.y * 4 + wave) * 32;

    const int frow = lane & 15;
    const int koff = (lane >> 4) << 4;

    const unsigned short* Bg  = BT + (size_t)ng * 32 * Kpad;
    const unsigned short* pa0 = A + (size_t)(mrow + frow) * Kpad + koff;
    const unsigned short* pa1 = pa0 + (size_t)16 * Kpad;

    if (wave == 0)
        tdm_load_b(Bg, (unsigned)(size_t)&sB[0][0], KC, Kpad);

    v8f acc00 = {}, acc01 = {}, acc10 = {}, acc11 = {};

    for (int c = 0; c < nChunk; ++c) {
        if (wave == 0) __builtin_amdgcn_s_wait_tensorcnt(0);
        __syncthreads();                                 // staged chunk visible to all
        if (wave == 0 && c + 1 < nChunk)
            tdm_load_b(Bg + (size_t)(c + 1) * KC,
                       (unsigned)(size_t)&sB[(c + 1) & 1][0], KC, Kpad);

        const unsigned short* pA0 = pa0 + c * KC;
        const unsigned short* pA1 = pa1 + c * KC;
        const unsigned short* lb0 = &sB[c & 1][0] + (size_t)frow * KC + koff;
        const unsigned short* lb1 = lb0 + (size_t)16 * KC;

        for (int ks = 0; ks < KC; ks += 32) {
            __builtin_prefetch(pA0 + ks + 64, 0, 0);
            __builtin_prefetch(pA1 + ks + 64, 0, 0);
            v16bf a0 = __builtin_bit_cast(v16bf, *(const v8u*)(pA0 + ks));
            v16bf a1 = __builtin_bit_cast(v16bf, *(const v8u*)(pA1 + ks));
            v16bf b0 = __builtin_bit_cast(v16bf, *(const v8u*)(lb0 + ks));
            v16bf b1 = __builtin_bit_cast(v16bf, *(const v8u*)(lb1 + ks));
            acc00 = wmma_bf16(a0, b0, acc00);
            acc01 = wmma_bf16(a0, b1, acc01);
            acc10 = wmma_bf16(a1, b0, acc10);
            acc11 = wmma_bf16(a1, b1, acc11);
        }
        __syncthreads();                                 // done reading this buffer
    }

    // epilogue: VGPR j -> row j (lanes 0-15) / j+8 (lanes 16-31), col = lane&15
    const int rb = (lane >> 4) << 3;
    auto store_tile = [&](const v8f& acc, int mi, int ni) {
        const int ct = ng * 32 + ni * 16;
        if (ct >= Nout) return;                          // padded N tile (uniform)
        const int col = ct + (lane & 15);
        const float bv = bias ? bias[col] : 0.f;
        float sc = 1.f, sh = 0.f;
        if (bnG) {
            float r = rsqrtf(bnV[col] + BN_EPS);
            sc = r * bnG[col];
            sh = bnB[col] - bnM[col] * sc;
        }
#pragma unroll
        for (int j = 0; j < 8; ++j) {
            float v = acc[j] + bv;
            if (bnG) v = sc * v + sh;
            if      (act == 1) v = v > 0.f ? v : expm1f(v);                       // ELU
            else if (act == 2) { v = expf(v); v = fminf(fmaxf(v, 1e-5f), 1e6f); } // mean
            else if (act == 3) { float sp = v > 20.f ? v : log1pf(expf(v));
                                 v = fminf(fmaxf(sp, 1e-4f), 1e4f); }             // disp
            else if (act == 4) v = 1.f / (1.f + expf(-v));                        // sigmoid
            C[(size_t)(mrow + mi * 16 + rb + j) * Nout + col] = v;
        }
    };
    store_tile(acc00, 0, 0); store_tile(acc01, 0, 1);
    store_tile(acc10, 1, 0); store_tile(acc11, 1, 1);
}

// ---------------- GAT attention kernels (H=3) ----------------

__global__ void k_elr(const float* __restrict__ h0, const float* __restrict__ al,
                      const float* __restrict__ ar, float* __restrict__ el,
                      float* __restrict__ er, int N, int F) {
    int i = blockIdx.x * blockDim.x + threadIdx.x;
    if (i >= N * 3) return;
    int n = i / 3, hh = i % 3;
    const float* row = h0 + (size_t)n * 3 * F + hh * F;
    const float* pal = al + hh * F;
    const float* par = ar + hh * F;
    float sl = 0.f, sr = 0.f;
    for (int f = 0; f < F; ++f) { float v = row[f]; sl += v * pal[f]; sr += v * par[f]; }
    el[i] = sl; er[i] = sr;
}

__global__ void k_edge_max(const int* __restrict__ src, const int* __restrict__ dst,
                           const float* __restrict__ el, const float* __restrict__ er,
                           float* __restrict__ ebuf, int* __restrict__ mord, int E) {
    int e = blockIdx.x * blockDim.x + threadIdx.x;
    if (e >= E) return;
    int s = src[e], d = dst[e];
    for (int h = 0; h < 3; ++h) {
        float v = el[s * 3 + h] + er[d * 3 + h];
        v = v > 0.f ? v : 0.2f * v;                      // leaky_relu(0.2)
        ebuf[(size_t)e * 3 + h] = v;
        atomicMax(&mord[d * 3 + h], fOrd(v));
    }
}

__global__ void k_edge_expsum(const int* __restrict__ dst, float* __restrict__ ebuf,
                              const int* __restrict__ mord, float* __restrict__ sbuf, int E) {
    int e = blockIdx.x * blockDim.x + threadIdx.x;
    if (e >= E) return;
    int d = dst[e];
    for (int h = 0; h < 3; ++h) {
        float m  = fUnord(mord[d * 3 + h]);
        float ee = expf(ebuf[(size_t)e * 3 + h] - m);
        ebuf[(size_t)e * 3 + h] = ee;
        atomicAdd(&sbuf[d * 3 + h], ee);
    }
}

__global__ void k_edge_agg(const int* __restrict__ src, const int* __restrict__ dst,
                           const float* __restrict__ ebuf, const float* __restrict__ sbuf,
                           const float* __restrict__ h0, float* __restrict__ agg, int F) {
    int e = blockIdx.x;
    int t = threadIdx.x;                // 0..3F-1
    int hh = t / F;
    int s = src[e], d = dst[e];
    float alpha = ebuf[(size_t)e * 3 + hh] / sbuf[d * 3 + hh];
    atomicAdd(&agg[(size_t)d * 3 * F + t], h0[(size_t)s * 3 * F + t] * alpha);
}

__global__ void k_gat_post(const float* __restrict__ agg, const float* __restrict__ bias,
                           float* __restrict__ y, int N, int F, int doElu,
                           const float* __restrict__ g, const float* __restrict__ bb,
                           const float* __restrict__ mm, const float* __restrict__ vv) {
    long i = (long)blockIdx.x * blockDim.x + threadIdx.x;
    if (i >= (long)N * F) return;
    int n = (int)(i / F), f = (int)(i % F);
    float acc = 0.f;
    for (int h = 0; h < 3; ++h) {
        float v = agg[(size_t)n * 3 * F + h * F + f] + bias[h * F + f];
        if (doElu) v = v > 0.f ? v : expm1f(v);
        acc += v;
    }
    if (g) {
        float a = rsqrtf(vv[f] + BN_EPS) * g[f];
        float c = bb[f] - mm[f] * a;
        acc = a * acc + c;      // bn applied twice (reference bug preserved)
        acc = a * acc + c;
    }
    y[(size_t)n * F + f] = acc;
}

// ---------------- fuse: LayerNorm + scSE + residual, one wave per node (Z=32) ----

__global__ void k_fuse(const float* __restrict__ z1, const float* __restrict__ z2,
                       const float* __restrict__ lnG, const float* __restrict__ lnB,
                       const float* __restrict__ fc1W, const float* __restrict__ fc1b,
                       const float* __restrict__ fc2W, const float* __restrict__ fc2b,
                       const float* __restrict__ sseW, const float* __restrict__ sseb,
                       float* __restrict__ outZ, int N) {
    int gtid = blockIdx.x * blockDim.x + threadIdx.x;
    int node = gtid >> 5;
    int lane = gtid & 31;
    if (node >= N) return;
    float a = z1[(size_t)node * 32 + lane];
    float b = z2[(size_t)node * 32 + lane];
    float mu1 = wave_sum32(a) * (1.f / 32.f);
    float d1  = a - mu1;
    float v1  = wave_sum32(d1 * d1) * (1.f / 32.f);
    float ln1 = d1 * rsqrtf(v1 + LN_EPS) * lnG[lane] + lnB[lane];
    float mu2 = wave_sum32(b) * (1.f / 32.f);
    float d2  = b - mu2;
    float v2  = wave_sum32(d2 * d2) * (1.f / 32.f);
    float ln2 = d2 * rsqrtf(v2 + LN_EPS) * lnG[lane] + lnB[lane];
    float p0 = wave_sum32(ln1) * (1.f / 32.f);
    float p1 = wave_sum32(ln2) * (1.f / 32.f);
    float u  = fmaxf(p0 * fc1W[0] + p1 * fc1W[1] + fc1b[0], 0.f);
    float c0 = 1.f / (1.f + expf(-(u * fc2W[0] + fc2b[0])));
    float c1 = 1.f / (1.f + expf(-(u * fc2W[1] + fc2b[1])));
    float s  = 1.f / (1.f + expf(-(ln1 * sseW[0] + ln2 * sseW[1] + sseb[0])));
    float o0 = THO_C * ln1 * c0 + ln1 * s + a;
    float o1 = THO_C * ln2 * c1 + ln2 * s + b;
    outZ[(size_t)node * 32 + lane] = o0 + o1;
}

// ---------------- host orchestration ----------------

extern "C" void kernel_launch(void* const* d_in, const int* in_sizes, int n_in,
                              void* d_out, int out_size, void* d_ws, size_t ws_size,
                              hipStream_t stream) {
    (void)n_in; (void)out_size; (void)ws_size;

    const int E = in_sizes[4];
    const float* x1 = (const float*)d_in[0];
    const float* x2 = (const float*)d_in[1];
    const float* n1 = (const float*)d_in[2];
    const float* n2 = (const float*)d_in[3];
    const int* src  = (const int*)d_in[4];
    const int* dst  = (const int*)d_in[5];

    auto F32 = [&](int i) { return (const float*)d_in[i]; };

    // flat param indices (depth-first dict order of setup_inputs)
    const int E1 = 6, E2 = 26;
    const int LNG = 46, LNB = 47;
    const int FC1W = 48, FC1B = 49, FC2W = 50, FC2B = 51, SSEW = 52, SSEB = 53;
    const int D1B = 54, D2B = 66;
    const int MEAN1W = 78, MEAN1B = 79, DISP1W = 80, DISP1B = 81, PI1W = 82, PI1B = 83;
    const int MEAN2W = 84, MEAN2B = 85, DISP2W = 86, DISP2B = 87, PI2W = 88, PI2B = 89;

    size_t off = 0;
    auto alloc = [&](size_t bytes) -> void* {
        void* p = (char*)d_ws + off;
        off = (off + bytes + 255) & ~(size_t)255;
        return p;
    };

    auto cvtW = [&](int idx, int K, int Nout, int Kpad) -> unsigned short* {
        int Npad32 = (Nout + 31) & ~31;
        unsigned short* o = (unsigned short*)alloc((size_t)Npad32 * Kpad * 2);
        long t = (long)Npad32 * Kpad;
        k_cvt_wT<<<(int)((t + 255) / 256), 256, 0, stream>>>(F32(idx), o, K, Nout, Kpad,
                                                             Npad32);
        return o;
    };

    unsigned short* wt0_1 = cvtW(E1 + 0, 2000, 768, 2016);
    unsigned short* wt0_2 = cvtW(E2 + 0, 2000, 768, 2016);
    unsigned short* wt1_1 = cvtW(E1 + 8, 256, 192, 256);
    unsigned short* wt1_2 = cvtW(E2 + 8, 256, 192, 256);
    unsigned short* wtm_1 = cvtW(E1 + 16, 64, 96, 64);
    unsigned short* wtm_2 = cvtW(E2 + 16, 64, 96, 64);
    unsigned short* wd1l0 = cvtW(D1B + 0, 32, 64, 32);
    unsigned short* wd1l1 = cvtW(D1B + 6, 64, 256, 64);
    unsigned short* wd2l0 = cvtW(D2B + 0, 32, 64, 32);
    unsigned short* wd2l1 = cvtW(D2B + 6, 64, 256, 64);
    unsigned short* wMe1  = cvtW(MEAN1W, 256, 2000, 256);
    unsigned short* wDi1  = cvtW(DISP1W, 256, 2000, 256);
    unsigned short* wPi1  = cvtW(PI1W,   256, 2000, 256);
    unsigned short* wMe2  = cvtW(MEAN2W, 256, 2000, 256);
    unsigned short* wDi2  = cvtW(DISP2W, 256, 2000, 256);
    unsigned short* wPi2  = cvtW(PI2W,   256, 2000, 256);

    unsigned short* XB = (unsigned short*)alloc((size_t)NN * 2016 * 2);
    float* HBUF  = (float*)alloc((size_t)NN * 768 * 4);
    float* AGG   = (float*)alloc((size_t)NN * 768 * 4);
    float* EL    = (float*)alloc((size_t)NN * 3 * 4);
    float* ER    = (float*)alloc((size_t)NN * 3 * 4);
    float* EBUF  = (float*)alloc((size_t)E * 3 * 4);
    int*   MORD  = (int*)  alloc((size_t)NN * 3 * 4);
    float* SBUF  = (float*)alloc((size_t)NN * 3 * 4);
    float* XMIDA = (float*)alloc((size_t)NN * 256 * 4);
    float* XMIDB = (float*)alloc((size_t)NN * 64 * 4);
    float* Z1    = (float*)alloc((size_t)NN * 32 * 4);
    float* Z2    = (float*)alloc((size_t)NN * 32 * 4);
    float* Z3    = (float*)alloc((size_t)NN * 32 * 4);
    float* Z4    = (float*)alloc((size_t)NN * 32 * 4);
    float* ZF    = (float*)alloc((size_t)NN * 32 * 4);
    float* M64   = (float*)alloc((size_t)NN * 64 * 4);
    float* H1    = (float*)alloc((size_t)NN * 256 * 4);

    float* out = (float*)d_out;
    const size_t NZ = (size_t)NN * 32, ND = (size_t)NN * 2000;
    float* oZ0  = out;
    float* oMe1 = out + NZ;
    float* oMe2 = oMe1 + ND;
    float* oDi1 = oMe2 + ND;
    float* oDi2 = oDi1 + ND;
    float* oPi1 = oDi2 + ND;
    float* oPi2 = oPi1 + ND;

    auto gemm = [&](const unsigned short* A, const unsigned short* BT, float* C,
                    int Nout, int Kpad, const float* bias,
                    const float* g, const float* b, const float* m, const float* v,
                    int act) {
        const int KC = (Kpad > 256) ? 224 : Kpad;       // 2016 = 9 * 224
        const int nChunk = Kpad / KC;
        const int Npad32 = (Nout + 31) & ~31;
        dim3 grid(Npad32 / 32, NN / 128);               // NN % 128 == 0
        k_gemm<<<grid, 128, 0, stream>>>(A, BT, C, Nout, Kpad, KC, nChunk,
                                         bias, g, b, m, v, act);
    };

    auto cvtA = [&](const float* X, const float* noise, float sigma, int K, int Kpad) {
        long t = (long)NN * Kpad;
        k_cvt_x<<<(int)((t + 255) / 256), 256, 0, stream>>>(X, noise, sigma, XB, NN, K, Kpad);
    };

    auto gat_layer = [&](int Kpad, const unsigned short* WT,
                         const float* al, const float* ar, const float* bias,
                         int F, int doElu,
                         const float* g, const float* bb, const float* mm, const float* vv,
                         float* Y) {
        const int Nout3 = 3 * F;
        gemm(XB, WT, HBUF, Nout3, Kpad, nullptr, nullptr, nullptr, nullptr, nullptr, 0);
        k_elr<<<(NN * 3 + 255) / 256, 256, 0, stream>>>(HBUF, al, ar, EL, ER, NN, F);
        k_fill_i32<<<(NN * 3 + 255) / 256, 256, 0, stream>>>(MORD, (int)0x80000000,
                                                             (long)NN * 3);
        k_fill_f32<<<(NN * 3 + 255) / 256, 256, 0, stream>>>(SBUF, 0.f, (long)NN * 3);
        long ta = (long)NN * Nout3;
        k_fill_f32<<<(int)((ta + 255) / 256), 256, 0, stream>>>(AGG, 0.f, ta);
        k_edge_max<<<(E + 255) / 256, 256, 0, stream>>>(src, dst, EL, ER, EBUF, MORD, E);
        k_edge_expsum<<<(E + 255) / 256, 256, 0, stream>>>(dst, EBUF, MORD, SBUF, E);
        k_edge_agg<<<E, Nout3, 0, stream>>>(src, dst, EBUF, SBUF, HBUF, AGG, F);
        long tp = (long)NN * F;
        k_gat_post<<<(int)((tp + 255) / 256), 256, 0, stream>>>(AGG, bias, Y, NN, F, doElu,
                                                                g, bb, mm, vv);
    };

    auto run_encoder = [&](const float* X, const float* noise, int eb,
                           const unsigned short* w0, const unsigned short* w1,
                           const unsigned short* wm, float* zOut) {
        cvtA(X, noise, SIGMA1, 2000, 2016);
        gat_layer(2016, w0, F32(eb + 1), F32(eb + 2), F32(eb + 3), 256, 1,
                  F32(eb + 4), F32(eb + 5), F32(eb + 6), F32(eb + 7), XMIDA);
        cvtA(XMIDA, nullptr, 0.f, 256, 256);
        gat_layer(256, w1, F32(eb + 9), F32(eb + 10), F32(eb + 11), 64, 1,
                  F32(eb + 12), F32(eb + 13), F32(eb + 14), F32(eb + 15), XMIDB);
        cvtA(XMIDB, nullptr, 0.f, 64, 64);
        gat_layer(64, wm, F32(eb + 17), F32(eb + 18), F32(eb + 19), 32, 0,
                  nullptr, nullptr, nullptr, nullptr, zOut);
    };

    auto fuse = [&](const float* a, const float* b, float* o) {
        k_fuse<<<(NN * 32) / 256, 256, 0, stream>>>(a, b, F32(LNG), F32(LNB),
                                                    F32(FC1W), F32(FC1B), F32(FC2W), F32(FC2B),
                                                    F32(SSEW), F32(SSEB), o, NN);
    };

    auto run_decoder = [&](int db, const unsigned short* wl0, const unsigned short* wl1,
                           const unsigned short* wM, const unsigned short* wD,
                           const unsigned short* wP, int bM, int bD, int bP,
                           float* oM, float* oD, float* oP) {
        cvtA(ZF, nullptr, 0.f, 32, 32);
        gemm(XB, wl0, M64, 64, 32, F32(db + 1),
             F32(db + 2), F32(db + 3), F32(db + 4), F32(db + 5), 1);
        cvtA(M64, nullptr, 0.f, 64, 64);
        gemm(XB, wl1, H1, 256, 64, F32(db + 7),
             F32(db + 8), F32(db + 9), F32(db + 10), F32(db + 11), 1);
        cvtA(H1, nullptr, 0.f, 256, 256);
        gemm(XB, wM, oM, 2000, 256, F32(bM), nullptr, nullptr, nullptr, nullptr, 2);
        gemm(XB, wD, oD, 2000, 256, F32(bD), nullptr, nullptr, nullptr, nullptr, 3);
        gemm(XB, wP, oP, 2000, 256, F32(bP), nullptr, nullptr, nullptr, nullptr, 4);
    };

    // ---- pipeline ----
    run_encoder(x1, n1, E1, wt0_1, wt1_1, wtm_1, Z1);      // noisy pass, modality 1
    run_encoder(x2, n2, E2, wt0_2, wt1_2, wtm_2, Z2);      // noisy pass, modality 2
    fuse(Z1, Z2, ZF);
    run_decoder(D1B, wd1l0, wd1l1, wMe1, wDi1, wPi1, MEAN1B, DISP1B, PI1B, oMe1, oDi1, oPi1);
    run_decoder(D2B, wd2l0, wd2l1, wMe2, wDi2, wPi2, MEAN2B, DISP2B, PI2B, oMe2, oDi2, oPi2);
    run_encoder(x1, nullptr, E1, wt0_1, wt1_1, wtm_1, Z3); // clean pass, modality 1
    run_encoder(x2, nullptr, E2, wt0_2, wt1_2, wtm_2, Z4); // clean pass, modality 2
    fuse(Z3, Z4, oZ0);                                     // z0 straight into d_out
}
